// ChamferDistanceLoss_16922171146733
// MI455X (gfx1250) — matrix operand order repaired and checked by reading, not compiled, not run on hardware
//
#include <hip/hip_runtime.h>
#include <hip/hip_bf16.h>

typedef __attribute__((ext_vector_type(16))) __bf16 v16bf;
typedef __attribute__((ext_vector_type(8)))  __bf16 v8bf;
typedef __attribute__((ext_vector_type(8)))  float  v8f;

#define B_   8
#define N_   4096
#define M_   4096
#define D_   128
#define TILE 128

#define FWD_COUNT (B_ * M_)                  // 32768 forward-min slots
#define NBLK ((N_ / TILE) * (M_ / TILE))     // 1024 blocks -> backward partials

// chunked bf16 layout: per (b, tile, khalf): [hi 128x64 | lo 128x64] = 16384 elems (32KB)
#define CHUNK_ELEMS 16384
#define LO_BYTE_OFF 16384                    // byte offset of lo block inside a chunk
#define BUF_ELEMS   32768                    // LDS buffer: X chunk + Y chunk (64KB)

// ---------------------------------------------------------------- async-to-LDS (probe-verified)
#if defined(__has_builtin)
#  if __has_builtin(__builtin_amdgcn_global_load_async_to_lds_b128)
#    define HAVE_ASYNC_LDS 1
#  endif
#endif

#ifdef HAVE_ASYNC_LDS
typedef int v4i_t __attribute__((vector_size(16)));
typedef __attribute__((address_space(1))) v4i_t GA_v4i;   // global (AS1)
typedef __attribute__((address_space(3))) v4i_t LA_v4i;   // LDS (AS3)
#endif

// 16B copy; imm offset O applies to BOTH global and LDS address (ISA INST_OFFSET semantics)
template<int O>
__device__ __forceinline__ void cp16(void* l, const void* g) {
#ifdef HAVE_ASYNC_LDS
    __builtin_amdgcn_global_load_async_to_lds_b128((GA_v4i*)g, (LA_v4i*)l, O, 0);
#else
    *(uint4*)((char*)l + O) = *(const uint4*)((const char*)g + O);
#endif
}
__device__ __forceinline__ void async_wait0() {
#ifdef HAVE_ASYNC_LDS
#if __has_builtin(__builtin_amdgcn_s_wait_asynccnt)
    __builtin_amdgcn_s_wait_asynccnt(0);
#else
    asm volatile("s_wait_asynccnt 0" ::: "memory");
#endif
#endif
}
// copy one 64B strip of a chunk (hi part + matching lo part) = 8 x 16B
__device__ __forceinline__ void cp_chunk_strip(void* l, const void* g) {
    cp16<0>(l, g);   cp16<16>(l, g);  cp16<32>(l, g);  cp16<48>(l, g);
    cp16<LO_BYTE_OFF +  0>(l, g); cp16<LO_BYTE_OFF + 16>(l, g);
    cp16<LO_BYTE_OFF + 32>(l, g); cp16<LO_BYTE_OFF + 48>(l, g);
}

// ---------------------------------------------------------------- init
__global__ void init_ws_kernel(unsigned* ws) {
    int i = blockIdx.x * blockDim.x + threadIdx.x;
    if (i < FWD_COUNT) ws[i] = 0x7F800000u;           // +inf bits
    else if (i < FWD_COUNT + NBLK) ws[i] = 0u;        // 0.0f
}

// ---------------------------------------------------------------- pre-convert: fp32 -> chunked bf16 hi/lo + row norms
__global__ __launch_bounds__(256)
void convert_kernel(const float* __restrict__ src, __bf16* __restrict__ dst,
                    float* __restrict__ norm, int rows)
{
    int r = blockIdx.x * blockDim.x + threadIdx.x;
    if (r >= rows) return;
    const int b  = r >> 12;        // 4096 rows per batch
    const int n  = r & 4095;
    const int t  = n >> 7;         // tile index
    const int ri = n & 127;        // row in tile
    const float* p = src + (size_t)r * D_;
    float s = 0.f;
#pragma unroll
    for (int h = 0; h < 2; ++h) {  // K-half
        __bf16* ph = dst + ((size_t)((b * 32 + t) * 2 + h)) * CHUNK_ELEMS + (size_t)ri * 64;
        __bf16* pl = ph + 8192;    // lo block (+16KB)
#pragma unroll 4
        for (int i = 0; i < 64; i += 4) {
            float4 v = *(const float4*)(p + h * 64 + i);
            const float f[4] = {v.x, v.y, v.z, v.w};
#pragma unroll
            for (int c = 0; c < 4; ++c) {
                float x = f[c];
                __bf16 hh = (__bf16)x;
                ph[i + c] = hh;
                pl[i + c] = (__bf16)(x - (float)hh);
                s += x * x;
            }
        }
    }
    norm[r] = s;
}

// ---------------------------------------------------------------- fragment helpers
__device__ __forceinline__ v16bf ld_b_frag(const __bf16* p) {
    union { v16bf v; v8bf h[2]; } u;
    u.h[0] = *(const v8bf*)(p);
    u.h[1] = *(const v8bf*)(p + 8);
    return u.v;
}
__device__ __forceinline__ v16bf ld_a_frag(const __bf16* p0, const __bf16* p1) {
    union { v16bf v; v8bf h[2]; } u;
    u.h[0] = *(const v8bf*)(p0);
    u.h[1] = *(const v8bf*)(p1);
    return u.v;
}

// ---------------------------------------------------------------- main tile kernel (double-buffered pipeline)
__global__ __launch_bounds__(256)
void chamfer_tiles_kernel(const __bf16* __restrict__ Xc, const __bf16* __restrict__ Yc,
                          const float* __restrict__ xnorm, const float* __restrict__ ynorm,
                          unsigned* __restrict__ fwdMin, float* __restrict__ bwdPart)
{
    extern __shared__ char smem[];
    __bf16* lds = (__bf16*)smem;                         // 2 x BUF_ELEMS (2 x 64KB)
    float*  sx2 = (float*)(smem + 2 * BUF_ELEMS * 2);    // [2][128] (by batch parity)
    float*  sy2 = sx2 + 2 * TILE;                        // [2][128]
    float*  red = sy2 + 2 * TILE;                        // [8]

    const int tid  = threadIdx.x;
    const int lane = tid & 31;
    const int wid  = tid >> 5;
    const int hsel = lane >> 4;
    const int l16  = lane & 15;

    const int mTile = blockIdx.x * TILE;
    const int nTile = blockIdx.y * TILE;

    const int waveRow = (wid >> 2) * 64;   // 2 row-strips
    const int waveCol = (wid & 3)  * 32;   // 4 col-strips

    // per-thread strip inside a chunk: row + half-row (32 elems = 64B)
    const int q0 = (tid >> 1) * 64 + (tid & 1) * 32;

    auto stage_tiles = [&](int s, int q) {
        const int b = s >> 1, kh = s & 1;
        const __bf16* gX = Xc + ((size_t)((b * 32 + blockIdx.y) * 2 + kh)) * CHUNK_ELEMS + q0;
        const __bf16* gY = Yc + ((size_t)((b * 32 + blockIdx.x) * 2 + kh)) * CHUNK_ELEMS + q0;
        __bf16* lX = lds + q * BUF_ELEMS + q0;
        __bf16* lY = lX + CHUNK_ELEMS;
        cp_chunk_strip(lX, gX);
        cp_chunk_strip(lY, gY);
    };
    auto stage_norms = [&](int b) {
        if (tid < TILE) {
            const int slot = (b & 1) * TILE;
            sx2[slot + tid] = xnorm[(size_t)b * N_ + nTile + tid];
            sy2[slot + tid] = ynorm[(size_t)b * M_ + mTile + tid];
        }
    };

    // min over batch per element (backward term)
    float bmin[4][2][8];
#pragma unroll
    for (int rs = 0; rs < 4; ++rs)
#pragma unroll
        for (int cs = 0; cs < 2; ++cs)
#pragma unroll
            for (int v = 0; v < 8; ++v) bmin[rs][cs][v] = __builtin_inff();

    // ---- prologue: stage (b=0, kh=0) + norms for b=0
    stage_tiles(0, 0);
    stage_norms(0);
    async_wait0();
    __syncthreads();

    v8f acc[4][2];

    for (int s = 0; s < 2 * B_; ++s) {
        const int b  = s >> 1;
        const int kh = s & 1;
        const int p  = s & 1;

        // ---- prefetch next stage into the other buffer (runs under compute)
        if (s + 1 < 2 * B_) {
            stage_tiles(s + 1, p ^ 1);
            if (kh == 1) stage_norms(b + 1);
        }

        const __bf16* sXhi = lds + p * BUF_ELEMS;
        const __bf16* sXlo = sXhi + 8192;
        const __bf16* sYhi = sXhi + CHUNK_ELEMS;
        const __bf16* sYlo = sYhi + 8192;

        if (kh == 0) {
#pragma unroll
            for (int rs = 0; rs < 4; ++rs)
#pragma unroll
                for (int cs = 0; cs < 2; ++cs)
                    acc[rs][cs] = (v8f){0.f,0.f,0.f,0.f,0.f,0.f,0.f,0.f};
        }

        // ---- 2 K-chunks of 32 from this K-half, split-bf16 3-pass
#pragma unroll
        for (int kc = 0; kc < 64; kc += 32) {
            v16bf bh[2], bl[2];
#pragma unroll
            for (int cs = 0; cs < 2; ++cs) {
                const int c   = waveCol + cs * 16 + l16;   // Y row = B column
                const int off = c * 64 + kc + hsel * 16;   // 16 contiguous K
                bh[cs] = ld_b_frag(sYhi + off);
                bl[cs] = ld_b_frag(sYlo + off);
            }
#pragma unroll
            for (int rs = 0; rs < 4; ++rs) {
                const int r    = waveRow + rs * 16 + l16;  // X row = A row
                const int aoff = r * 64 + kc + hsel * 16;
                v16bf ah = ld_a_frag(sXhi + aoff, sXhi + aoff + 8);
                v16bf al = ld_a_frag(sXlo + aoff, sXlo + aoff + 8);
#pragma unroll
                for (int cs = 0; cs < 2; ++cs) {
                    acc[rs][cs] = __builtin_amdgcn_wmma_f32_16x16x32_bf16(
                        false, ah, false, bh[cs], (short)0, acc[rs][cs], false, false);
                    acc[rs][cs] = __builtin_amdgcn_wmma_f32_16x16x32_bf16(
                        false, ah, false, bl[cs], (short)0, acc[rs][cs], false, false);
                    acc[rs][cs] = __builtin_amdgcn_wmma_f32_16x16x32_bf16(
                        false, al, false, bh[cs], (short)0, acc[rs][cs], false, false);
                }
            }
        }

        // ---- epilogue once per batch (after second K-half)
        if (kh == 1) {
            const int slot = (b & 1) * TILE;
            float cmin0 = __builtin_inff(), cmin1 = __builtin_inff();
#pragma unroll
            for (int rs = 0; rs < 4; ++rs)
#pragma unroll
                for (int cs = 0; cs < 2; ++cs) {
                    const int nl  = waveRow + rs * 16 + hsel * 8;
                    const int ml  = waveCol + cs * 16 + l16;
                    const float y2v = sy2[slot + ml];
#pragma unroll
                    for (int v = 0; v < 8; ++v) {
                        float sq = sx2[slot + nl + v] + y2v - 2.f * acc[rs][cs][v];
                        float d  = sqrtf(fmaxf(sq, 0.f));
                        bmin[rs][cs][v] = fminf(bmin[rs][cs][v], d);
                        if (cs == 0) cmin0 = fminf(cmin0, d);
                        else         cmin1 = fminf(cmin1, d);
                    }
                }
            // combine the two lane-halves (rows 0-7/8-15 of each strip share columns)
            cmin0 = fminf(cmin0, __shfl_xor(cmin0, 16, 32));
            cmin1 = fminf(cmin1, __shfl_xor(cmin1, 16, 32));
            if (hsel == 0) {   // distances >= 0 -> uint order == float order
                atomicMin(&fwdMin[(size_t)b * M_ + mTile + waveCol +      l16], __float_as_uint(cmin0));
                atomicMin(&fwdMin[(size_t)b * M_ + mTile + waveCol + 16 + l16], __float_as_uint(cmin1));
            }
        }

        async_wait0();     // next buffer fully staged
        __syncthreads();   // all waves done with buffer p; prefetched data visible
    }

    // ---- backward partial: sum of per-element batch-mins over this tile
    float s = 0.f;
#pragma unroll
    for (int rs = 0; rs < 4; ++rs)
#pragma unroll
        for (int cs = 0; cs < 2; ++cs)
#pragma unroll
            for (int v = 0; v < 8; ++v) s += bmin[rs][cs][v];
#pragma unroll
    for (int off = 16; off >= 1; off >>= 1) s += __shfl_xor(s, off, 32);
    if (lane == 0) red[wid] = s;
    __syncthreads();
    if (tid == 0) {
        float t = 0.f;
        for (int w = 0; w < 8; ++w) t += red[w];
        bwdPart[blockIdx.y * gridDim.x + blockIdx.x] = t;
    }
}

// ---------------------------------------------------------------- finalize (deterministic order)
__global__ __launch_bounds__(256)
void finalize_kernel(const unsigned* __restrict__ fwdMin,
                     const float* __restrict__ bwdPart, float* __restrict__ out)
{
    __shared__ float red[256];
    const int tid = threadIdx.x;
    float fs = 0.f;
    for (int i = tid; i < FWD_COUNT; i += 256) fs += __uint_as_float(fwdMin[i]);
    float bs = 0.f;
    for (int i = tid; i < NBLK; i += 256) bs += bwdPart[i];
    red[tid] = fs * (1.0f / (float)FWD_COUNT) + bs * (1.0f / ((float)N_ * (float)M_));
    __syncthreads();
    for (int stride = 128; stride > 0; stride >>= 1) {
        if (tid < stride) red[tid] += red[tid + stride];
        __syncthreads();
    }
    if (tid == 0) out[0] = red[0];
}

// ---------------------------------------------------------------- launch
extern "C" void kernel_launch(void* const* d_in, const int* in_sizes, int n_in,
                              void* d_out, int out_size, void* d_ws, size_t ws_size,
                              hipStream_t stream) {
    (void)in_sizes; (void)n_in; (void)out_size; (void)ws_size;
    const float* X = (const float*)d_in[0];   // predicted_set (B,N,D) f32
    const float* Y = (const float*)d_in[1];   // target_set    (B,M,D) f32
    float* out = (float*)d_out;

    // ws layout
    unsigned* fwdMin  = (unsigned*)d_ws;                     // [32768] u32 (f32 bits)
    float*    bwdPart = (float*)d_ws + FWD_COUNT;            // [1024]  f32
    float*    xnorm   = (float*)d_ws + FWD_COUNT + NBLK;     // [B*N]   f32
    float*    ynorm   = xnorm + (size_t)B_ * N_;             // [B*M]   f32
    __bf16*   Xc      = (__bf16*)(ynorm + (size_t)B_ * M_);  // chunked hi/lo, 16MB
    __bf16*   Yc      = Xc + (size_t)B_ * 32 * 2 * CHUNK_ELEMS;

    init_ws_kernel<<<(FWD_COUNT + NBLK + 255) / 256, 256, 0, stream>>>(fwdMin);

    convert_kernel<<<(B_ * N_) / 256, 256, 0, stream>>>(X, Xc, xnorm, B_ * N_);
    convert_kernel<<<(B_ * M_) / 256, 256, 0, stream>>>(Y, Yc, ynorm, B_ * M_);

    const size_t shmem = (size_t)2 * BUF_ELEMS * sizeof(__bf16)   // double-buffered tiles (128KB)
                       + (size_t)2 * TILE * sizeof(float)         // sx2[2][128]
                       + (size_t)2 * TILE * sizeof(float)         // sy2[2][128]
                       + 8 * sizeof(float);                       // red
    dim3 grid(M_ / TILE, N_ / TILE);
    chamfer_tiles_kernel<<<grid, 256, shmem, stream>>>(Xc, Yc, xnorm, ynorm, fwdMin, bwdPart);

    finalize_kernel<<<1, 256, 0, stream>>>(fwdMin, bwdPart, out);
}
